// CosformerAttention_6691559047669
// MI455X (gfx1250) — matrix operand order repaired and checked by reading, not compiled, not run on hardware
//
#include <hip/hip_runtime.h>

// ---------------- feature probes ----------------
#if defined(__has_builtin)
# if __has_builtin(__builtin_amdgcn_global_load_async_to_lds_b128)
#  define HAVE_ASYNC_COPY 1
# endif
# if __has_builtin(__builtin_amdgcn_tensor_load_to_lds)
#  define HAVE_TDM 1
# endif
#endif

// ---------------- types ----------------
typedef __attribute__((ext_vector_type(16))) __bf16 v16bf;
typedef __attribute__((ext_vector_type(8)))  __bf16 v8bf;
typedef __attribute__((ext_vector_type(8)))  float  v8f;
typedef __attribute__((ext_vector_type(4)))  unsigned int u32x4;
typedef __attribute__((ext_vector_type(2)))  unsigned int u32x2;
typedef __attribute__((ext_vector_type(4)))  unsigned int tdm_u32x4;
typedef __attribute__((ext_vector_type(4)))  int          tdm_i32x4;
typedef __attribute__((ext_vector_type(8)))  int          tdm_i32x8;

// async-copy builtin wants GCC-vector int4 pointers with explicit address spaces
typedef int vsi4 __attribute__((vector_size(16)));
typedef vsi4 __attribute__((address_space(1))) * as1_i4p;
typedef vsi4 __attribute__((address_space(3))) * as3_i4p;

#define PI_HALF 1.57079632679489662f
#define LSEQ 2048
#define NB   8
#define EDIM 1024

__device__ __forceinline__ unsigned short f2bf(float x) {
    union { float f; unsigned u; } v; v.f = x;
    unsigned r = v.u + 0x7FFFu + ((v.u >> 16) & 1u);   // RNE
    return (unsigned short)(r >> 16);
}
__device__ __forceinline__ float bf2f(unsigned short h) {
    union { unsigned u; float f; } v; v.u = ((unsigned)h) << 16; return v.f;
}
__device__ __forceinline__ v8f vzero8() {
    v8f z;
#pragma unroll
    for (int i = 0; i < 8; i++) z[i] = 0.0f;
    return z;
}
__device__ __forceinline__ v16bf mk16(v8bf lo, v8bf hi) {
    v16bf r;
#pragma unroll
    for (int i = 0; i < 8; i++) { r[i] = lo[i]; r[i + 8] = hi[i]; }
    return r;
}
__device__ __forceinline__ v16bf ld_frag(const unsigned short* p0, const unsigned short* p1) {
    v8bf lo = *(const v8bf*)p0;
    v8bf hi = *(const v8bf*)p1;
    return mk16(lo, hi);
}
__device__ __forceinline__ void wait_async0() {
#if defined(__has_builtin) && __has_builtin(__builtin_amdgcn_s_wait_asynccnt)
    __builtin_amdgcn_s_wait_asynccnt(0);
#else
    asm volatile("s_wait_asynccnt 0" ::: "memory");
#endif
}
__device__ __forceinline__ void wait_tensor0() {
#if defined(__has_builtin) && __has_builtin(__builtin_amdgcn_s_wait_tensorcnt)
    __builtin_amdgcn_s_wait_tensorcnt((short)0);
#else
    asm volatile("s_wait_tensorcnt 0" ::: "memory");
#endif
}
#if defined(HAVE_ASYNC_COPY)
// global (AS1) -> LDS (AS3) 16-byte async copy; LDS offset = low 32 bits of generic addr
__device__ __forceinline__ void async_copy_b128(const void* gp, void* lp) {
    __builtin_amdgcn_global_load_async_to_lds_b128(
        (as1_i4p)(uintptr_t)gp,
        (as3_i4p)(unsigned)(uintptr_t)lp,
        0, 0);
}
#endif

// ---------------- shared GEMM body:  C(16384x1024) = A(16384x1024) * W(1024x1024)^T ----------------
// Block tile 128x256, 8 waves of 64x64 each, K staged in chunks of 32 (one bf16 WMMA deep).
template <bool ABF, bool OBF>
__device__ __forceinline__ void gemm_body(const void* Aop, const float* W, void* Cop, bool relu) {
    constexpr int Kd = 1024, Nd = 1024;
    __shared__ unsigned short As[128 * 32];   // [m][k] bf16
    __shared__ unsigned short Bs[256 * 32];   // [n][k] bf16 (== W rows)

    const int tid  = threadIdx.x;
    const int m0   = blockIdx.y * 128;
    const int n0   = blockIdx.x * 256;
    const int lane = tid & 31, wave = tid >> 5;
    const int wm = wave >> 2, wn = wave & 3;          // 2x4 wave grid
    const int lg = lane >> 4, lr = lane & 15;

    v8f acc[4][4];
#pragma unroll
    for (int mi = 0; mi < 4; mi++)
#pragma unroll
        for (int ni = 0; ni < 4; ni++) acc[mi][ni] = vzero8();

    for (int k0 = 0; k0 < Kd; k0 += 32) {
        // ---- stage A tile (128x32) ----
        if constexpr (ABF) {
            const unsigned short* Ab = (const unsigned short*)Aop;
#pragma unroll
            for (int i = 0; i < 2; i++) {
                int c = tid + 256 * i;               // 512 chunks of 8 bf16
                int row = c >> 2, off = (c & 3) * 8;
                const unsigned short* gp = Ab + (size_t)(m0 + row) * Kd + k0 + off;
                unsigned short* lp = As + row * 32 + off;
#if defined(HAVE_ASYNC_COPY)
                async_copy_b128(gp, lp);
#else
                *(u32x4*)lp = *(const u32x4*)gp;
#endif
            }
        } else {
            const float* Af = (const float*)Aop;
#pragma unroll
            for (int i = 0; i < 4; i++) {
                int f = tid + 256 * i;               // 1024 float4 chunks
                int row = f >> 3, c4 = (f & 7) * 4;
                const float* p = Af + (size_t)(m0 + row) * Kd + k0 + c4;
                unsigned short h0 = f2bf(p[0]), h1 = f2bf(p[1]), h2 = f2bf(p[2]), h3 = f2bf(p[3]);
                u32x2 pk; pk[0] = h0 | ((unsigned)h1 << 16); pk[1] = h2 | ((unsigned)h3 << 16);
                *(u32x2*)(As + row * 32 + c4) = pk;
            }
        }
        // ---- stage B tile (256x32) from W (always fp32) ----
#pragma unroll
        for (int i = 0; i < 8; i++) {
            int f = tid + 256 * i;
            int row = f >> 3, c4 = (f & 7) * 4;
            const float* p = W + (size_t)(n0 + row) * Kd + k0 + c4;
            unsigned short h0 = f2bf(p[0]), h1 = f2bf(p[1]), h2 = f2bf(p[2]), h3 = f2bf(p[3]);
            u32x2 pk; pk[0] = h0 | ((unsigned)h1 << 16); pk[1] = h2 | ((unsigned)h3 << 16);
            *(u32x2*)(Bs + row * 32 + c4) = pk;
            if (k0 + 32 < Kd) __builtin_prefetch(p + 32, 0, 1);   // next K-chunk of W
        }
        if constexpr (ABF) {
#if defined(HAVE_ASYNC_COPY)
            wait_async0();
#endif
        }
        __syncthreads();

        // ---- fragments + 16 WMMAs ----
        v16bf afr[4], bfr[4];
#pragma unroll
        for (int mi = 0; mi < 4; mi++) {
            const unsigned short* pr = As + (wm * 64 + mi * 16 + lr) * 32;
            afr[mi] = ld_frag(pr + lg * 8, pr + 16 + lg * 8);
        }
#pragma unroll
        for (int ni = 0; ni < 4; ni++) {
            const unsigned short* pr = Bs + (wn * 64 + ni * 16 + lr) * 32;
            bfr[ni] = ld_frag(pr + lg * 8, pr + 16 + lg * 8);
        }
#pragma unroll
        for (int mi = 0; mi < 4; mi++)
#pragma unroll
            for (int ni = 0; ni < 4; ni++)
                acc[mi][ni] = __builtin_amdgcn_wmma_f32_16x16x32_bf16(
                    false, afr[mi], false, bfr[ni], (short)0, acc[mi][ni], false, false);
        __syncthreads();
    }

    // ---- writeout ----
#pragma unroll
    for (int mi = 0; mi < 4; mi++) {
        int rbase = m0 + wm * 64 + mi * 16 + lg * 8;
#pragma unroll
        for (int ni = 0; ni < 4; ni++) {
            int col = n0 + wn * 64 + ni * 16 + lr;
#pragma unroll
            for (int r = 0; r < 8; r++) {
                float v = acc[mi][ni][r];
                if (relu) v = fmaxf(v, 0.0f);
                size_t idx = (size_t)(rbase + r) * Nd + col;
                if constexpr (OBF) ((unsigned short*)Cop)[idx] = f2bf(v);
                else               ((float*)Cop)[idx] = v;
            }
        }
    }
}

// ---------------- kernel 1: fused QKV projection (grid.z selects W / relu) ----------------
__global__ __launch_bounds__(256) void qkv_gemm(const float* X,
                                                const float* Wq, const float* Wk, const float* Wv,
                                                unsigned short* qb, unsigned short* kb, unsigned short* vb) {
    int z = blockIdx.z;
    const float* W = (z == 0) ? Wq : (z == 1) ? Wk : Wv;
    unsigned short* O = (z == 0) ? qb : (z == 1) ? kb : vb;
    gemm_body<false, true>(X, W, O, z < 2);
}

// ---------------- kernel 4: output projection ----------------
__global__ __launch_bounds__(256) void out_gemm(const unsigned short* A, const float* Wo, float* C) {
    gemm_body<true, false>(A, Wo, C, false);
}

// ---------------- kernel 2: per-head S1/S2 (64x64) + t1/t2 reduction over L ----------------
__global__ __launch_bounds__(256) void kv_head(const unsigned short* kb, const unsigned short* vb,
                                               unsigned short* Sbuf, float* tbuf) {
    __shared__ unsigned short Kt[64 * 32];        // k^T : [d][l]
    __shared__ unsigned short V2[2 * 64 * 32];    // [which][m][l]  (sin*v, cos*v)
    __shared__ float scs[32], scc[32];
    __shared__ float tred[2 * 4 * 64];

    const int tid = threadIdx.x;
    const int head = blockIdx.x;
    const int nb = head >> 4, h = head & 15;
    const int lane = tid & 31, wave = tid >> 5;
    const int which = wave >> 2, wn = wave & 3;   // waves 0-3 -> S1(sin), 4-7 -> S2(cos)
    const int lg = lane >> 4, lr = lane & 15;
    const int td = tid & 63, tg = tid >> 6;

    v8f acc[4];
#pragma unroll
    for (int mi = 0; mi < 4; mi++) acc[mi] = vzero8();
    float t1p = 0.0f, t2p = 0.0f;

    for (int lc = 0; lc < 64; lc++) {
        const int l0 = lc * 32;
        if (tid < 32) {
            float idx = PI_HALF * (float)(l0 + tid + 1) / (float)LSEQ;
            scs[tid] = __sinf(idx); scc[tid] = __cosf(idx);
        }
        __syncthreads();
#pragma unroll
        for (int i = 0; i < 8; i++) {
            int e = tid + 256 * i;                // 32 l x 64 d
            int l = e >> 6, d = e & 63;
            size_t g = ((size_t)(l0 + l) * NB + nb) * EDIM + h * 64 + d;
            Kt[d * 32 + l] = kb[g];
            float vv = bf2f(vb[g]);
            V2[0 * 2048 + d * 32 + l] = f2bf(scs[l] * vv);
            V2[1 * 2048 + d * 32 + l] = f2bf(scc[l] * vv);
        }
        __syncthreads();

        // t1/t2 partials (each thread owns (d=td, l stride group tg))
#pragma unroll
        for (int ls = 0; ls < 32; ls += 4) {
            int l = ls + tg;
            float kv = bf2f(Kt[td * 32 + l]);
            t1p += scs[l] * kv;
            t2p += scc[l] * kv;
        }

        // WMMA: A = k^T (64 x 32), B = scaled v (32 x 16 slice)
        v16bf afr[4];
#pragma unroll
        for (int mi = 0; mi < 4; mi++) {
            const unsigned short* pr = Kt + (mi * 16 + lr) * 32;
            afr[mi] = ld_frag(pr + lg * 8, pr + 16 + lg * 8);
        }
        const unsigned short* Bp = V2 + which * 2048 + (wn * 16 + lr) * 32;
        v16bf bfr = ld_frag(Bp + lg * 8, Bp + 16 + lg * 8);
#pragma unroll
        for (int mi = 0; mi < 4; mi++)
            acc[mi] = __builtin_amdgcn_wmma_f32_16x16x32_bf16(
                false, afr[mi], false, bfr, (short)0, acc[mi], false, false);
        __syncthreads();
    }

    // write S (bf16): Sbuf[head][which][d][m]
#pragma unroll
    for (int mi = 0; mi < 4; mi++) {
        int drow = mi * 16 + lg * 8;
        int m = wn * 16 + lr;
#pragma unroll
        for (int r = 0; r < 8; r++)
            Sbuf[((size_t)(head * 2 + which) * 64 + drow + r) * 64 + m] = f2bf(acc[mi][r]);
    }
    // reduce t partials
    tred[(0 * 4 + tg) * 64 + td] = t1p;
    tred[(1 * 4 + tg) * 64 + td] = t2p;
    __syncthreads();
    if (tid < 128) {
        int w = tid >> 6, d = tid & 63;
        float s = 0.0f;
#pragma unroll
        for (int g = 0; g < 4; g++) s += tred[(w * 4 + g) * 64 + d];
        tbuf[(size_t)(head * 2 + w) * 64 + d] = s;
    }
}

// ---------------- kernel 3: attn = z * (sin*q@S1 + cos*q@S2), bf16 out ----------------
__global__ __launch_bounds__(256) void attn_apply(const unsigned short* qb, const unsigned short* Sbuf,
                                                  const float* tbuf, unsigned short* attnb) {
    __shared__ unsigned short Qs[128 * 64];       // [l][d]
    __shared__ unsigned short S1t[64 * 64];       // [m][d]
    __shared__ unsigned short S2t[64 * 64];
    __shared__ float t1s[64], t2s[64], zs[128];

    const int tid = threadIdx.x;
    const int l0 = blockIdx.x * 128;
    const int head = blockIdx.y;
    const int nb = head >> 4, h = head & 15;

#if defined(HAVE_TDM)
    // Tensor Data Mover: one descriptor pulls the whole 128x64 bf16 Q tile into LDS.
    if (tid < 32) {
        const unsigned short* gsrc = qb + ((size_t)l0 * NB + nb) * EDIM + h * 64;
        unsigned long long ga = (unsigned long long)(uintptr_t)gsrc;
        unsigned ldsoff = (unsigned)(uintptr_t)&Qs[0];
        tdm_u32x4 g0;
        g0[0] = 1u;                                        // count=1, user mode
        g0[1] = ldsoff;                                    // lds_addr (bytes)
        g0[2] = (unsigned)ga;                              // global_addr[31:0]
        g0[3] = (unsigned)(ga >> 32) | 0x80000000u;        // global_addr[56:32] | type=2
        tdm_i32x8 g1;
        g1[0] = 0x10000;                                   // data_size=1 (2 bytes)
        g1[1] = 64 << 16;                                  // tensor_dim0 = 64
        g1[2] = 0;                                         // tensor_dim1 = 0x20000 (low16=0)
        g1[3] = 2 | (64 << 16);                            // tensor_dim1 hi | tile_dim0=64
        g1[4] = 128;                                       // tile_dim1=128, tile_dim2=0
        g1[5] = NB * EDIM;                                 // tensor_dim0_stride = 8192
        g1[6] = 0; g1[7] = 0;
        tdm_i32x4 z4 = {0, 0, 0, 0};
#if __clang_major__ >= 23
        tdm_i32x8 z8 = {0, 0, 0, 0, 0, 0, 0, 0};
        __builtin_amdgcn_tensor_load_to_lds(g0, g1, z4, z4, z8, 0);
#else
        __builtin_amdgcn_tensor_load_to_lds(g0, g1, z4, z4, 0);
#endif
        wait_tensor0();
    }
#else
#pragma unroll
    for (int i = 0; i < 4; i++) {                 // q tile: 128 x 64 bf16
        int c = tid + 256 * i;
        int row = c >> 3, off = (c & 7) * 8;
        size_t g = ((size_t)(l0 + row) * NB + nb) * EDIM + h * 64 + off;
        *(u32x4*)(Qs + row * 64 + off) = *(const u32x4*)(qb + g);
    }
#endif
#pragma unroll
    for (int i = 0; i < 16; i++) {                // S1/S2 transposed into [m][d]
        int e = tid + 256 * i;
        int d = e >> 6, m = e & 63;
        S1t[m * 64 + d] = Sbuf[((size_t)(head * 2 + 0) * 64 + d) * 64 + m];
        S2t[m * 64 + d] = Sbuf[((size_t)(head * 2 + 1) * 64 + d) * 64 + m];
    }
    if (tid < 64) {
        t1s[tid] = tbuf[(size_t)(head * 2 + 0) * 64 + tid];
        t2s[tid] = tbuf[(size_t)(head * 2 + 1) * 64 + tid];
    }
    __syncthreads();

    if (tid < 128) {                              // z per row
        float d1 = 0.0f, d2 = 0.0f;
#pragma unroll
        for (int d = 0; d < 64; d++) {
            float qv = bf2f(Qs[tid * 64 + d]);
            d1 += qv * t1s[d]; d2 += qv * t2s[d];
        }
        float idx = PI_HALF * (float)(l0 + tid + 1) / (float)LSEQ;
        float den = __sinf(idx) * d1 + __cosf(idx) * d2;
        zs[tid] = 1.0f / fmaxf(den, 1e-6f);
    }
    __syncthreads();

    const int lane = tid & 31, wave = tid >> 5;
    const int lg = lane >> 4, lr = lane & 15;
    v8f a1[4], a2[4];
#pragma unroll
    for (int ni = 0; ni < 4; ni++) { a1[ni] = vzero8(); a2[ni] = vzero8(); }

#pragma unroll
    for (int kc = 0; kc < 64; kc += 32) {
        const unsigned short* pq = Qs + (wave * 16 + lr) * 64 + kc;
        v16bf af = ld_frag(pq + lg * 8, pq + 16 + lg * 8);
#pragma unroll
        for (int ni = 0; ni < 4; ni++) {
            const unsigned short* p1 = S1t + (ni * 16 + lr) * 64 + kc;
            const unsigned short* p2 = S2t + (ni * 16 + lr) * 64 + kc;
            v16bf b1 = ld_frag(p1 + lg * 8, p1 + 16 + lg * 8);
            v16bf b2 = ld_frag(p2 + lg * 8, p2 + 16 + lg * 8);
            a1[ni] = __builtin_amdgcn_wmma_f32_16x16x32_bf16(false, af, false, b1, (short)0, a1[ni], false, false);
            a2[ni] = __builtin_amdgcn_wmma_f32_16x16x32_bf16(false, af, false, b2, (short)0, a2[ni], false, false);
        }
    }

    float sv[8], cv[8], zv[8];
#pragma unroll
    for (int r = 0; r < 8; r++) {
        int rowl = wave * 16 + lg * 8 + r;
        float idx = PI_HALF * (float)(l0 + rowl + 1) / (float)LSEQ;
        sv[r] = __sinf(idx); cv[r] = __cosf(idx); zv[r] = zs[rowl];
    }
#pragma unroll
    for (int ni = 0; ni < 4; ni++) {
        int m = ni * 16 + lr;
#pragma unroll
        for (int r = 0; r < 8; r++) {
            int rowl = wave * 16 + lg * 8 + r;
            float val = (sv[r] * a1[ni][r] + cv[r] * a2[ni][r]) * zv[r];
            attnb[((size_t)(l0 + rowl) * NB + nb) * EDIM + h * 64 + m] = f2bf(val);
        }
    }
}

// ---------------- launch ----------------
extern "C" void kernel_launch(void* const* d_in, const int* in_sizes, int n_in,
                              void* d_out, int out_size, void* d_ws, size_t ws_size,
                              hipStream_t stream) {
    const float* query = (const float*)d_in[0];
    const float* Wq = (const float*)d_in[1];
    const float* Wk = (const float*)d_in[2];
    const float* Wv = (const float*)d_in[3];
    const float* Wo = (const float*)d_in[4];

    char* ws = (char*)d_ws;
    const size_t SZ = (size_t)16384 * 1024;           // elements of one (L*N, E) buffer
    unsigned short* qb    = (unsigned short*)(ws);
    unsigned short* kb    = (unsigned short*)(ws + SZ * 2);
    unsigned short* vb    = (unsigned short*)(ws + SZ * 4);
    unsigned short* attnb = (unsigned short*)(ws + SZ * 6);
    unsigned short* Sbuf  = (unsigned short*)(ws + SZ * 8);
    float*          tbuf  = (float*)(ws + SZ * 8 + (size_t)128 * 2 * 64 * 64 * 2);

    qkv_gemm<<<dim3(4, 128, 3), 256, 0, stream>>>(query, Wq, Wk, Wv, qb, kb, vb);
    kv_head<<<dim3(128), 256, 0, stream>>>(kb, vb, Sbuf, tbuf);
    attn_apply<<<dim3(16, 128), 256, 0, stream>>>(qb, Sbuf, tbuf, attnb);
    out_gemm<<<dim3(4, 128), 256, 0, stream>>>(attnb, Wo, (float*)d_out);
}
